// DeformConvAdd2d_15719580304167
// MI455X (gfx1250) — matrix operand order repaired and checked by reading, not compiled, not run on hardware
//
#include <hip/hip_runtime.h>
#include <hip/hip_bf16.h>

// Deformable conv2d (3x3, pad 1) + residual add, MI455X / gfx1250.
// GEMM part done with v_wmma_f32_16x16x32_f16 (f16 inputs, f32 accum).
// B=4, C=64, H=W=128, Cout=64, KK=9.

typedef __attribute__((ext_vector_type(16))) _Float16 v16h;
typedef __attribute__((ext_vector_type(8)))  float    v8f;

#define H_ 128
#define W_ 128
#define HW 16384
#define C_ 64
#define CO 64
#define NT 64            // spatial tile (N) per workgroup
#define PITCH 66         // LDS row pitch in halves (odd dword stride -> no bank conflicts)

union Frag {
    v16h v;
    unsigned int u[8];
};

__global__ __launch_bounds__(256)
void deform_conv_wmma(const float* __restrict__ x,
                      const float* __restrict__ offset,
                      const float* __restrict__ x2,
                      const float* __restrict__ weight,
                      const float* __restrict__ bias,
                      float* __restrict__ out)
{
    __shared__ _Float16 w_chunk[CO * PITCH];     // [o][c]  for current tap
    __shared__ _Float16 cols_chunk[NT * PITCH];  // [n][c]  for current tap

    const int tid = threadIdx.x;
    const int b   = blockIdx.y;
    const int tile_base = blockIdx.x * NT;       // start index in flattened H*W

    // ---- build-phase thread mapping: 4 threads per n, 16 channels each ----
    const int bn = tid & 63;                     // n within tile
    const int cq = tid >> 6;                     // channel quarter 0..3
    const int n_global = tile_base + bn;
    const int hh = n_global >> 7;                // /W
    const int ww = n_global & 127;               // %W

    // ---- WMMA-phase thread mapping ----
    const int lane    = tid & 31;
    const int wv      = tid >> 5;                // wave 0..7
    const int m_base  = (wv >> 1) * 16;          // 4 M-tiles
    const int nt0     = (wv & 1) * 2;            // 2 N-subtiles per wave
    const int lane_lo = lane & 15;
    const int lane_hi = (lane >> 4) & 1;

    const int arow  = (m_base + lane_lo) * PITCH;
    const int brow0 = (nt0 * 16 + lane_lo) * PITCH;
    const int brow1 = ((nt0 + 1) * 16 + lane_lo) * PITCH;

    v8f acc0 = {};
    v8f acc1 = {};

    const float* xb = x + (size_t)(b * C_ + cq * 16) * HW;

    for (int kk = 0; kk < 9; ++kk) {
        __syncthreads();   // protect LDS against previous tap's fragment reads

        // ---- stage weight chunk: w_chunk[o][c] = weight[o][c][kk] ----
        for (int i = tid; i < CO * C_; i += 256) {
            const int o = i >> 6;
            const int c = i & 63;
            w_chunk[o * PITCH + c] = (_Float16)weight[(o * C_ + c) * 9 + kk];
        }

        // ---- build bilinear cols chunk for this tap ----
        {
            const int kyi = kk / 3, kxi = kk % 3;
            const float oy = offset[((b * 18 + 2 * kk    ) * H_ + hh) * W_ + ww];
            const float ox = offset[((b * 18 + 2 * kk + 1) * H_ + hh) * W_ + ww];
            const float py = (float)(hh - 1 + kyi) + oy;
            const float px = (float)(ww - 1 + kxi) + ox;
            const float fy = floorf(py), fx = floorf(px);
            const int   y0 = (int)fy,   x0 = (int)fx;
            const float ly = py - fy,   lx = px - fx;

            const int vy0 = (y0 >= 0) & (y0 < H_);
            const int vy1 = (y0 + 1 >= 0) & (y0 + 1 < H_);
            const int vx0 = (x0 >= 0) & (x0 < W_);
            const int vx1 = (x0 + 1 >= 0) & (x0 + 1 < W_);

            const float w00 = (vy0 & vx0) ? (1.f - ly) * (1.f - lx) : 0.f;
            const float w01 = (vy0 & vx1) ? (1.f - ly) * lx         : 0.f;
            const float w10 = (vy1 & vx0) ? ly * (1.f - lx)         : 0.f;
            const float w11 = (vy1 & vx1) ? ly * lx                 : 0.f;

            const int y0c = min(max(y0, 0), H_ - 1);
            const int y1c = min(max(y0 + 1, 0), H_ - 1);
            const int x0c = min(max(x0, 0), W_ - 1);
            const int x1c = min(max(x0 + 1, 0), W_ - 1);

            const int i00 = y0c * W_ + x0c;
            const int i01 = y0c * W_ + x1c;
            const int i10 = y1c * W_ + x0c;
            const int i11 = y1c * W_ + x1c;

            _Float16* dst = &cols_chunk[bn * PITCH + cq * 16];
            #pragma unroll
            for (int cc = 0; cc < 16; ++cc) {
                const float* xc = xb + cc * HW;
                const float val = w00 * xc[i00] + w01 * xc[i01]
                                + w10 * xc[i10] + w11 * xc[i11];
                dst[cc] = (_Float16)val;
            }
        }

        __syncthreads();

        // ---- WMMA accumulate over this tap's K-chunk of 64 (2 K-steps of 32) ----
        #pragma unroll
        for (int k0 = 0; k0 < 64; k0 += 32) {
            Frag a, b0, b1;
            #pragma unroll
            for (int j = 0; j < 8; ++j) {
                const int h0 = 2 * j;
                // A (16x32 f16, ISA layout): K = h + (h&8) + 8*lane_hi
                const int ka = k0 + h0 + (h0 & 8) + lane_hi * 8;
                a.u[j]  = *(const unsigned int*)&w_chunk[arow + ka];
                // B (32x16 f16): K = h + (h&8) + 8*lane_hi
                const int kb = k0 + h0 + (h0 & 8) + lane_hi * 8;
                b0.u[j] = *(const unsigned int*)&cols_chunk[brow0 + kb];
                b1.u[j] = *(const unsigned int*)&cols_chunk[brow1 + kb];
            }
            acc0 = __builtin_amdgcn_wmma_f32_16x16x32_f16(
                       false, a.v, false, b0.v, (short)0, acc0, false, false);
            acc1 = __builtin_amdgcn_wmma_f32_16x16x32_f16(
                       false, a.v, false, b1.v, (short)0, acc1, false, false);
        }
    }

    // ---- epilogue: bias + residual, coalesced along N ----
    #pragma unroll
    for (int v = 0; v < 8; ++v) {
        const int M = m_base + v + lane_hi * 8;          // C/D layout (ISA 7.12.2)
        const int N0 = tile_base + nt0 * 16 + lane_lo;
        const size_t idx0 = (size_t)(b * CO + M) * HW + N0;
        const float bz = bias[M];
        out[idx0]      = acc0[v] + bz + x2[idx0];
        out[idx0 + 16] = acc1[v] + bz + x2[idx0 + 16];
    }
}

extern "C" void kernel_launch(void* const* d_in, const int* in_sizes, int n_in,
                              void* d_out, int out_size, void* d_ws, size_t ws_size,
                              hipStream_t stream) {
    const float* x      = (const float*)d_in[0];
    const float* offset = (const float*)d_in[1];
    const float* x2     = (const float*)d_in[2];
    const float* weight = (const float*)d_in[3];
    const float* bias   = (const float*)d_in[4];
    float* out = (float*)d_out;

    dim3 grid(HW / NT, 4);   // 256 spatial tiles x 4 batches
    dim3 block(256);         // 8 wave32
    deform_conv_wmma<<<grid, block, 0, stream>>>(x, offset, x2, weight, bias, out);
}
